// LSTMbyHand_74629351735855
// MI455X (gfx1250) — matrix operand order repaired and checked by reading, not compiled, not run on hardware
//
#include <hip/hip_runtime.h>

// Model dims (fixed by the reference)
#define BB 16
#define SS 256
#define VV 32000
#define HH 128
#define LL 2
#define MROWS (BB*SS)          // 4096
#define NTILES_OUT (VV/16)     // 2000
#define MTILES (MROWS/16)      // 256
#define KSTEPS (HH/32)         // 4
#define NGATE (3*HH)           // 384
#define NTILES_GATE (NGATE/16) // 24

typedef __attribute__((ext_vector_type(16))) _Float16 v16h;
typedef __attribute__((ext_vector_type(8)))  float    v8f;

// ---------------------------------------------------------------------------
// A-fragment (16x32 f16, M x K) scatter index for writer threads.
// Storage layout: [mtile][kstep][lane][16 halfs], so a wave's fragment load is
// one contiguous 32B read per lane (global_load_b128 x2).
// A layout (contiguous K, matches B): lane = half*16 + m; element e holds
// K = ks*32 + 16*half + e.
// ---------------------------------------------------------------------------
__device__ __forceinline__ size_t afrag_index(int r, int h) {
    int mtile  = r >> 4;
    int mm     = r & 15;
    int ks     = h >> 5;
    int kk     = h & 31;
    int g      = kk >> 4;          // which 16-wide K group
    int kk16   = kk & 15;
    int half   = kk >> 4;          // lane half
    int j      = (kk16 & 7) >> 1;  // VGPR within group
    int parity = kk & 1;
    int i      = g * 4 + j;
    int lane   = half * 16 + mm;
    return ((((size_t)mtile * KSTEPS + ks) * 32 + lane) << 4) + kk16;
}

__device__ __forceinline__ float sigmoidf_(float z) {
    return 1.0f / (1.0f + __expf(-z));
}

// ---------------------------------------------------------------------------
// One-time weight converts (run each launch; deterministic).
// B-fragment layout (32x16 f16, K x N): lane = half*16 + n; element e holds
// K = ks*32 + 16*half + e  (linear in e -> simple convert).
// ---------------------------------------------------------------------------
__global__ void mlstm_convert_gates(const float* __restrict__ wlr,
                                    const float* __restrict__ wpr,
                                    const float* __restrict__ wcs,
                                    _Float16* __restrict__ WgFrag) {
    int t = blockIdx.x * blockDim.x + threadIdx.x;   // [l][nt][ks][lane]
    if (t >= LL * NTILES_GATE * KSTEPS * 32) return;
    int lane = t & 31;
    int ks   = (t >> 5) & (KSTEPS - 1);
    int nt   = (t >> 7) % NTILES_GATE;
    int l    = t / (32 * KSTEPS * NTILES_GATE);
    int n    = nt * 16 + (lane & 15);
    int gate = n >> 7;
    int col  = n & (HH - 1);
    const float* w = (gate == 0) ? wlr : ((gate == 1) ? wpr : wcs);
    int kb = ks * 32 + ((lane >> 4) << 4);
    _Float16* dst = WgFrag + ((size_t)t << 4);
#pragma unroll
    for (int e = 0; e < 16; ++e)
        dst[e] = (_Float16)w[((size_t)l * HH + (kb + e)) * HH + col];
}

__global__ void mlstm_convert_fcw(const float* __restrict__ fcw,
                                  _Float16* __restrict__ Ff) {
    int t = blockIdx.x * blockDim.x + threadIdx.x;   // [nt][ks][lane]
    if (t >= NTILES_OUT * KSTEPS * 32) return;
    int lane = t & 31;
    int ks   = (t >> 5) & (KSTEPS - 1);
    int nt   = t >> 7;
    int n    = nt * 16 + (lane & 15);
    int kb   = ks * 32 + ((lane >> 4) << 4);
    _Float16* dst = Ff + ((size_t)t << 4);
#pragma unroll
    for (int e = 0; e < 16; ++e)
        dst[e] = (_Float16)fcw[(size_t)(kb + e) * VV + n];
}

// Embedding gather -> f16 A-fragment layout
__global__ void mlstm_gather(const int* __restrict__ seq,
                             const float* __restrict__ emb,
                             _Float16* __restrict__ X0) {
    int idx = blockIdx.x * blockDim.x + threadIdx.x;  // 0 .. 4096*128-1
    int r = idx >> 7;
    int h = idx & (HH - 1);
    int tok = seq[r];
    X0[afrag_index(r, h)] = (_Float16)emb[(size_t)tok * HH + h];
}

// ---------------------------------------------------------------------------
// Gate GEMM: G[4096][384] = X @ Wcat  (f16 WMMA, fp32 accumulate)
// grid (24, 64), block 128 (4 waves); wave -> one 16x16 tile
// ---------------------------------------------------------------------------
__global__ void mlstm_gemm_gates(const _Float16* __restrict__ Xfrag,
                                 const _Float16* __restrict__ WgFrag,
                                 float* __restrict__ G) {
    int lane  = threadIdx.x & 31;
    int w     = threadIdx.x >> 5;
    int ntile = blockIdx.x;
    int mtile = blockIdx.y * 4 + w;
    v8f c = {};
#pragma unroll
    for (int ks = 0; ks < KSTEPS; ++ks) {
        v16h a = *(const v16h*)(Xfrag  + ((((size_t)mtile * KSTEPS + ks) * 32 + lane) << 4));
        v16h b = *(const v16h*)(WgFrag + ((((size_t)ntile * KSTEPS + ks) * 32 + lane) << 4));
        c = __builtin_amdgcn_wmma_f32_16x16x32_f16(false, a, false, b, (short)0, c, false, false);
    }
    int half = lane >> 4;
    int n    = ntile * 16 + (lane & 15);
#pragma unroll
    for (int i = 0; i < 8; ++i) {
        int row = mtile * 16 + i + 8 * half;
        G[(size_t)row * NGATE + n] = c[i];
    }
}

// ---------------------------------------------------------------------------
// Parallel activation kernel: pull all transcendentals OUT of the serial
// recurrence.  x_t = a_t * x_{t-1} + d_t  with
//   a = sigmoid(glr + blr),  d = sigmoid(gpr + bpr) * tanh(gcs + bcs)
// AD[r][h] = float2(a, d)  (coalesced: consecutive h -> consecutive float2)
// ---------------------------------------------------------------------------
__global__ void mlstm_act(const float* __restrict__ G,
                          const float* __restrict__ blr,
                          const float* __restrict__ bpr,
                          const float* __restrict__ bcs,
                          float2* __restrict__ AD) {
    int idx = blockIdx.x * blockDim.x + threadIdx.x;  // 0 .. 4096*128-1
    int r = idx >> 7;
    int h = idx & (HH - 1);
    const float* g = G + (size_t)r * NGATE;
    float a = sigmoidf_(g[h] + blr[h]);
    float d = sigmoidf_(g[HH + h] + bpr[h]) * tanhf(g[2 * HH + h] + bcs[h]);
    float2 v; v.x = a; v.y = d;
    AD[idx] = v;
}

// ---------------------------------------------------------------------------
// Linear time-scan: x_t = a_t * x_{t-1} + d_t over S=256 steps, 2048 channels.
// Loads batched 8 steps ahead (addresses independent of state) so L2 latency
// is amortized; the dependent chain is pure FMA.
// Writes layer output as f16 A-fragments.
// ---------------------------------------------------------------------------
__global__ void mlstm_scan(const float2* __restrict__ AD,
                           _Float16* __restrict__ Xout) {
    int tid = blockIdx.x * blockDim.x + threadIdx.x;  // 0..2047
    int b = tid >> 7;
    int h = tid & (HH - 1);
    const float2* p = AD + ((size_t)(b << 8) * HH + h);  // step stride = HH float2
    float x = 0.0f;
    for (int s0 = 0; s0 < SS; s0 += 8) {
        float2 v[8];
#pragma unroll
        for (int u = 0; u < 8; ++u)
            v[u] = p[(size_t)(s0 + u) * HH];
#pragma unroll
        for (int u = 0; u < 8; ++u) {
            x = fmaf(v[u].x, x, v[u].y);
            Xout[afrag_index((b << 8) + s0 + u, h)] = (_Float16)x;
        }
    }
}

// ---------------------------------------------------------------------------
// Output GEMM: out[4096][32000] = X2 @ fc_w + fc_b  (memory-bound: 524 MB NT
// stores). grid (125, 256), block 256 (8 waves); wave -> two 16x16 N-tiles.
// ---------------------------------------------------------------------------
__global__ void mlstm_gemm_out(const _Float16* __restrict__ Xfrag,
                               const _Float16* __restrict__ Ff,
                               const float* __restrict__ fc_b,
                               float* __restrict__ out) {
    int lane  = threadIdx.x & 31;
    int w     = threadIdx.x >> 5;        // 0..7
    int mtile = blockIdx.y;              // 0..255
    int nt    = blockIdx.x * 16 + w * 2; // even tile index
    v8f c0 = {}, c1 = {};
#pragma unroll
    for (int ks = 0; ks < KSTEPS; ++ks) {
        v16h a  = *(const v16h*)(Xfrag + ((((size_t)mtile * KSTEPS + ks) * 32 + lane) << 4));
        v16h b0 = *(const v16h*)(Ff + ((((size_t)nt       * KSTEPS + ks) * 32 + lane) << 4));
        v16h b1 = *(const v16h*)(Ff + ((((size_t)(nt + 1) * KSTEPS + ks) * 32 + lane) << 4));
        c0 = __builtin_amdgcn_wmma_f32_16x16x32_f16(false, a, false, b0, (short)0, c0, false, false);
        c1 = __builtin_amdgcn_wmma_f32_16x16x32_f16(false, a, false, b1, (short)0, c1, false, false);
    }
    int half = lane >> 4;
    int n0   = nt * 16 + (lane & 15);
    float bias0 = fc_b[n0];
    float bias1 = fc_b[n0 + 16];
#pragma unroll
    for (int i = 0; i < 8; ++i) {
        int row = mtile * 16 + i + 8 * half;
        float* p = out + (size_t)row * VV + n0;
        __builtin_nontemporal_store(c0[i] + bias0, p);
        __builtin_nontemporal_store(c1[i] + bias1, p + 16);
    }
}

// ---------------------------------------------------------------------------
extern "C" void kernel_launch(void* const* d_in, const int* in_sizes, int n_in,
                              void* d_out, int out_size, void* d_ws, size_t ws_size,
                              hipStream_t stream) {
    const int*   seq  = (const int*)d_in[0];
    const float* emb  = (const float*)d_in[1];
    const float* wlr  = (const float*)d_in[2];
    const float* wpr  = (const float*)d_in[3];
    const float* wcs  = (const float*)d_in[4];
    const float* blr  = (const float*)d_in[5];
    const float* bpr  = (const float*)d_in[6];
    const float* bcs  = (const float*)d_in[7];
    const float* fcw  = (const float*)d_in[8];
    const float* fcb  = (const float*)d_in[9];
    float* out = (float*)d_out;

    char* ws = (char*)d_ws;
    // workspace layout (all power-of-two-ish aligned), total ~22 MB
    _Float16* WgFrag = (_Float16*)(ws + 0);           // 2*24*4*32*16 halfs = 192 KB
    _Float16* Ff     = (_Float16*)(ws + 196608);      // 2000*4*32*16 halfs = 8 MB
    _Float16* X0     = (_Float16*)(ws + 8388608);     // 1 MB
    _Float16* X1     = (_Float16*)(ws + 9437184);     // 1 MB
    _Float16* X2     = (_Float16*)(ws + 10485760);    // 1 MB
    float*    G      = (float*)   (ws + 11534336);    // 4096*384*4 = 6 MB
    float2*   AD     = (float2*)  (ws + 17825792);    // 4096*128*8 = 4 MB

    const size_t wg_layer_stride = (size_t)NTILES_GATE * KSTEPS * 32 * 16;

    // weight converts + gather (independent, all on stream)
    mlstm_convert_gates<<<24, 256, 0, stream>>>(wlr, wpr, wcs, WgFrag);
    mlstm_convert_fcw<<<1000, 256, 0, stream>>>(fcw, Ff);
    mlstm_gather<<<2048, 256, 0, stream>>>(seq, emb, X0);

    // layer 0: parallel gate GEMM -> parallel activations -> linear scan
    mlstm_gemm_gates<<<dim3(24, 64), 128, 0, stream>>>(X0, WgFrag, G);
    mlstm_act<<<2048, 256, 0, stream>>>(G, blr, bpr, bcs, AD);
    mlstm_scan<<<8, 256, 0, stream>>>(AD, X1);

    // layer 1
    mlstm_gemm_gates<<<dim3(24, 64), 128, 0, stream>>>(X1, WgFrag + wg_layer_stride, G);
    mlstm_act<<<2048, 256, 0, stream>>>(G, blr + HH, bpr + HH, bcs + HH, AD);
    mlstm_scan<<<8, 256, 0, stream>>>(AD, X2);

    // output projection (memory-bound, nontemporal stores)
    mlstm_gemm_out<<<dim3(125, 256), 256, 0, stream>>>(X2, Ff, fcb, out);
}
